// GCN_85529978733394
// MI455X (gfx1250) — matrix operand order repaired and checked by reading, not compiled, not run on hardware
//
#include <hip/hip_runtime.h>
#include <stdint.h>

#define NNODES 100000
#define NEDGES 3200000

typedef __attribute__((ext_vector_type(16))) __bf16          v16bf;
typedef __attribute__((ext_vector_type(8)))  float           v8f;
typedef __attribute__((ext_vector_type(8)))  unsigned short  u16x8;
typedef __attribute__((ext_vector_type(16))) unsigned short  u16x16;
typedef __attribute__((ext_vector_type(4)))  int             i32x4;

typedef __attribute__((address_space(1))) i32x4  g_i32x4;    // global
typedef __attribute__((address_space(3))) i32x4  l_i32x4;    // LDS

__device__ __forceinline__ float bf2f(unsigned short h) {
    return __uint_as_float(((unsigned int)h) << 16);
}
__device__ __forceinline__ unsigned short f2bf(float f) {
    unsigned int u = __float_as_uint(f);
    u += 0x7FFFu + ((u >> 16) & 1u);          // round-to-nearest-even
    return (unsigned short)(u >> 16);
}

__device__ __forceinline__ void wait_async0() {
#if __has_builtin(__builtin_amdgcn_s_wait_asynccnt)
    __builtin_amdgcn_s_wait_asynccnt(0);
#else
    asm volatile("s_wait_asynccnt 0x0" ::: "memory");
#endif
}

// ---------------------------------------------------------------- degrees
__global__ __launch_bounds__(256) void k_degrees(const int* __restrict__ src,
                                                 const int* __restrict__ dst,
                                                 int* __restrict__ degIn,
                                                 int* __restrict__ degOut) {
    int e = blockIdx.x * 256 + threadIdx.x;
    if (e >= NEDGES) return;
    atomicAdd(&degIn[dst[e]], 1);
    atomicAdd(&degOut[src[e]], 1);
}

__global__ __launch_bounds__(256) void k_norms(const int* __restrict__ degIn,
                                               const int* __restrict__ degOut,
                                               float* __restrict__ normSrc,
                                               float* __restrict__ normDst) {
    int i = blockIdx.x * 256 + threadIdx.x;
    if (i >= NNODES) return;
    normSrc[i] = rsqrtf(fmaxf((float)degOut[i], 1.0f));
    normDst[i] = rsqrtf(fmaxf((float)degIn[i], 1.0f));
}

// ---------------------------------------------------------------- scan (CSR offsets)
__global__ __launch_bounds__(256) void k_block_sums(const int* __restrict__ deg,
                                                    int* __restrict__ blkSums, int n) {
    __shared__ int s[256];
    int i = blockIdx.x * 256 + threadIdx.x;
    s[threadIdx.x] = (i < n) ? deg[i] : 0;
    __syncthreads();
    for (int o = 128; o > 0; o >>= 1) {
        if (threadIdx.x < o) s[threadIdx.x] += s[threadIdx.x + o];
        __syncthreads();
    }
    if (threadIdx.x == 0) blkSums[blockIdx.x] = s[0];
}

__global__ __launch_bounds__(512) void k_scan_sums(int* __restrict__ blkSums, int nb) {
    __shared__ int s[512];
    int t = threadIdx.x;
    s[t] = (t < nb) ? blkSums[t] : 0;
    __syncthreads();
    for (int o = 1; o < 512; o <<= 1) {
        int x = (t >= o) ? s[t - o] : 0;
        __syncthreads();
        s[t] += x;
        __syncthreads();
    }
    if (t < nb) blkSums[t] = (t == 0) ? 0 : s[t - 1];   // exclusive
}

__global__ __launch_bounds__(256) void k_scan_final(const int* __restrict__ deg,
                                                    const int* __restrict__ blkSums,
                                                    int* __restrict__ offsets, int n) {
    __shared__ int s[256];
    int i = blockIdx.x * 256 + threadIdx.x;
    int v = (i < n) ? deg[i] : 0;
    s[threadIdx.x] = v;
    __syncthreads();
    for (int o = 1; o < 256; o <<= 1) {
        int x = (threadIdx.x >= o) ? s[threadIdx.x - o] : 0;
        __syncthreads();
        s[threadIdx.x] += x;
        __syncthreads();
    }
    if (i < n) offsets[i + 1] = blkSums[blockIdx.x] + s[threadIdx.x];
    if (i == 0) offsets[0] = 0;
}

__global__ __launch_bounds__(256) void k_copy_i32(const int* __restrict__ a,
                                                  int* __restrict__ b, int n) {
    int i = blockIdx.x * 256 + threadIdx.x;
    if (i < n) b[i] = a[i];
}

__global__ __launch_bounds__(256) void k_fill_csr(const int* __restrict__ src,
                                                  const int* __restrict__ dst,
                                                  int* __restrict__ cursor,
                                                  int* __restrict__ csr) {
    int e = blockIdx.x * 256 + threadIdx.x;
    if (e >= NEDGES) return;
    int slot = atomicAdd(&cursor[dst[e]], 1);
    csr[slot] = src[e];
}

// ---------------------------------------------------------------- weight transpose fp32 -> bf16 (Wt[n][k] = W[k][n])
__global__ __launch_bounds__(256) void k_transpose(const float* __restrict__ W,
                                                   unsigned short* __restrict__ Wt,
                                                   int K, int Nout) {
    int idx = blockIdx.x * 256 + threadIdx.x;
    if (idx >= K * Nout) return;
    int k = idx / Nout, n = idx % Nout;
    Wt[(size_t)n * K + k] = f2bf(W[idx]);
}

// ---------------------------------------------------------------- layer 1: t[i,j] = bf16(deg_in[i]*norm_src[i]*W1[j])
__global__ __launch_bounds__(256) void k_layer1(const int* __restrict__ degIn,
                                                const float* __restrict__ normSrc,
                                                const float* __restrict__ W1,
                                                unsigned short* __restrict__ t) {
    size_t idx = (size_t)blockIdx.x * 256 + threadIdx.x;
    if (idx >= (size_t)NNODES * 512) return;
    int i = (int)(idx >> 9);
    int j = (int)(idx & 511);
    t[idx] = f2bf((float)degIn[i] * normSrc[i] * W1[j]);
}

// ---------------------------------------------------------------- WMMA bf16 GEMM: T = (A @ Wt^T) * norm_src, out bf16
// Block owns one 64-col strip; weight panel staged in LDS via async copy.
// Each wave computes a 32x64 output tile (2 M-tiles x 4 N-tiles = 8 WMMAs/k-step).
// A: N x K bf16, Bt: Nout x K bf16 (= W^T), T: N x Nout bf16
__global__ __launch_bounds__(256) void k_gemm(const unsigned short* __restrict__ A,
                                              const unsigned short* __restrict__ Bt,
                                              const float* __restrict__ normSrc,
                                              unsigned short* __restrict__ T,
                                              int K, int Nout, int nPairs,
                                              int blocksPerStrip) {
    __shared__ unsigned short sB[64 * 512];        // 64 KB max (K<=512)

    const int strip = blockIdx.x / blocksPerStrip; // 64-wide output strip
    const int pb    = blockIdx.x % blocksPerStrip;
    const int n0    = strip * 64;

    // ---- stage weight panel Bt[n0 .. n0+63][0..K) into LDS (contiguous 64*K*2 bytes)
    {
        const char* g = (const char*)(Bt + (size_t)n0 * K);
        char*       l = (char*)sB;
        const int nvec = (64 * K * 2) / 16;        // 16B chunks
        for (int i = threadIdx.x; i < nvec; i += 256) {
#if __has_builtin(__builtin_amdgcn_global_load_async_to_lds_b128)
            __builtin_amdgcn_global_load_async_to_lds_b128(
                (g_i32x4*)(g + (size_t)i * 16),
                (l_i32x4*)(l + (size_t)i * 16), 0, 0);
#else
            *(u16x8*)(l + (size_t)i * 16) = *(const u16x8*)(g + (size_t)i * 16);
#endif
        }
#if __has_builtin(__builtin_amdgcn_global_load_async_to_lds_b128)
        wait_async0();
#endif
        __syncthreads();
    }

    const int lane = threadIdx.x & 31;
    const int pair = pb * 8 + (threadIdx.x >> 5);  // 32-row M-pair handled by this wave
    if (pair >= nPairs) return;                    // wave-uniform (after barrier)
    const int m0 = pair * 32;

    const int row = lane & 15;
    const int kbA = (lane >> 4) << 4;              // A frag: 0 or 16  (contiguous k, matches B)
    const int kbB = (lane >> 4) << 4;              // B frag: 0 or 16  (contiguous k, matches A)

    const unsigned short* ap0 = A + (size_t)(m0 + row) * K;
    const unsigned short* ap1 = A + (size_t)(m0 + 16 + row) * K;
    const unsigned short* lb0 = sB + (size_t)(0  + row) * K;
    const unsigned short* lb1 = sB + (size_t)(16 + row) * K;
    const unsigned short* lb2 = sB + (size_t)(32 + row) * K;
    const unsigned short* lb3 = sB + (size_t)(48 + row) * K;

    v8f accA0 = {}, accA1 = {}, accA2 = {}, accA3 = {};   // rows m0..m0+15
    v8f accB0 = {}, accB1 = {}, accB2 = {}, accB3 = {};   // rows m0+16..m0+31

    for (int k = 0; k < K; k += 32) {
        __builtin_prefetch(ap0 + k + 128, 0, 1);
        __builtin_prefetch(ap1 + k + 128, 0, 1);

        union { u16x8 u[2]; v16bf v; } a0, a1;
        a0.u[0] = *(const u16x8*)(ap0 + k + kbA);
        a0.u[1] = *(const u16x8*)(ap0 + k + 8 + kbA);
        a1.u[0] = *(const u16x8*)(ap1 + k + kbA);
        a1.u[1] = *(const u16x8*)(ap1 + k + 8 + kbA);

        union { u16x16 u; v16bf v; } b0, b1, b2, b3;
        b0.u = *(const u16x16*)(lb0 + k + kbB);
        b1.u = *(const u16x16*)(lb1 + k + kbB);
        b2.u = *(const u16x16*)(lb2 + k + kbB);
        b3.u = *(const u16x16*)(lb3 + k + kbB);

        accA0 = __builtin_amdgcn_wmma_f32_16x16x32_bf16(false, a0.v, false, b0.v, (short)0, accA0, false, false);
        accB0 = __builtin_amdgcn_wmma_f32_16x16x32_bf16(false, a1.v, false, b0.v, (short)0, accB0, false, false);
        accA1 = __builtin_amdgcn_wmma_f32_16x16x32_bf16(false, a0.v, false, b1.v, (short)0, accA1, false, false);
        accB1 = __builtin_amdgcn_wmma_f32_16x16x32_bf16(false, a1.v, false, b1.v, (short)0, accB1, false, false);
        accA2 = __builtin_amdgcn_wmma_f32_16x16x32_bf16(false, a0.v, false, b2.v, (short)0, accA2, false, false);
        accB2 = __builtin_amdgcn_wmma_f32_16x16x32_bf16(false, a1.v, false, b2.v, (short)0, accB2, false, false);
        accA3 = __builtin_amdgcn_wmma_f32_16x16x32_bf16(false, a0.v, false, b3.v, (short)0, accA3, false, false);
        accB3 = __builtin_amdgcn_wmma_f32_16x16x32_bf16(false, a1.v, false, b3.v, (short)0, accB3, false, false);
    }

    // D layout: VGPR r -> row base + 8*(lane>>4) + r, col n0 + (lane&15)
    const int cBase  = n0 + (lane & 15);
    const int rBase0 = m0 + ((lane >> 4) << 3);
    const int rBase1 = rBase0 + 16;
    float ns0[8], ns1[8];
#pragma unroll
    for (int r = 0; r < 8; ++r) { ns0[r] = normSrc[rBase0 + r]; ns1[r] = normSrc[rBase1 + r]; }
#pragma unroll
    for (int r = 0; r < 8; ++r) {
        size_t ro0 = (size_t)(rBase0 + r) * Nout + cBase;
        size_t ro1 = (size_t)(rBase1 + r) * Nout + cBase;
        T[ro0 +  0] = f2bf(accA0[r] * ns0[r]);
        T[ro0 + 16] = f2bf(accA1[r] * ns0[r]);
        T[ro0 + 32] = f2bf(accA2[r] * ns0[r]);
        T[ro0 + 48] = f2bf(accA3[r] * ns0[r]);
        T[ro1 +  0] = f2bf(accB0[r] * ns1[r]);
        T[ro1 + 16] = f2bf(accB1[r] * ns1[r]);
        T[ro1 + 32] = f2bf(accB2[r] * ns1[r]);
        T[ro1 + 48] = f2bf(accB3[r] * ns1[r]);
    }
}

// ---------------------------------------------------------------- CSR gather-aggregate + norm_dst + bias + relu -> bf16
__global__ __launch_bounds__(256) void k_aggregate(const unsigned short* __restrict__ t,
                                                   const int* __restrict__ csr,
                                                   const int* __restrict__ off,
                                                   const float* __restrict__ normDst,
                                                   const float* __restrict__ bias,
                                                   unsigned short* __restrict__ hout,
                                                   int F) {
    int node = blockIdx.x;
    __shared__ int so[2];
    if (threadIdx.x < 2) so[threadIdx.x] = off[node + threadIdx.x];
    __syncthreads();
    int j = threadIdx.x * 2;
    if (j >= F) return;                            // F multiple of 64 -> wave-uniform
    float a0 = 0.f, a1 = 0.f;
    int e0 = so[0], e1 = so[1];
    for (int e = e0; e < e1; ++e) {
        const unsigned short* rowp = t + (size_t)csr[e] * F;
        unsigned int w = *(const unsigned int*)(rowp + j);
        a0 += bf2f((unsigned short)(w & 0xFFFFu));
        a1 += bf2f((unsigned short)(w >> 16));
    }
    float nd = normDst[node];
    float r0 = fmaxf(a0 * nd + bias[j], 0.f);
    float r1 = fmaxf(a1 * nd + bias[j + 1], 0.f);
    *(unsigned int*)(hout + (size_t)node * F + j) =
        (unsigned int)f2bf(r0) | ((unsigned int)f2bf(r1) << 16);
}

// ---------------------------------------------------------------- mean pool + classifier
__global__ __launch_bounds__(128) void k_pool(const unsigned short* __restrict__ h,
                                              float* __restrict__ pooled) {
    int j = threadIdx.x;                           // 128 features
    float acc = 0.f;
    for (int i = blockIdx.x; i < NNODES; i += gridDim.x)
        acc += bf2f(h[(size_t)i * 128 + j]);
    atomicAdd(&pooled[j], acc);
}

__global__ __launch_bounds__(32) void k_classify(const float* __restrict__ pooled,
                                                 const float* __restrict__ Wc,
                                                 const float* __restrict__ bc,
                                                 float* __restrict__ out) {
    int c = threadIdx.x;
    if (c >= 10) return;
    float s = 0.f;
    for (int jj = 0; jj < 128; ++jj) s += pooled[jj] * Wc[jj * 10 + c];
    out[c] = s * (1.0f / NNODES) + bc[c];
}

// ---------------------------------------------------------------- launch
extern "C" void kernel_launch(void* const* d_in, const int* in_sizes, int n_in,
                              void* d_out, int out_size, void* d_ws, size_t ws_size,
                              hipStream_t stream) {
    (void)in_sizes; (void)n_in; (void)out_size; (void)ws_size;
    const int*   src = (const int*)d_in[0];
    const int*   dst = (const int*)d_in[1];
    const float* W1  = (const float*)d_in[2];
    const float* b1  = (const float*)d_in[3];
    const float* W2  = (const float*)d_in[4];
    const float* b2  = (const float*)d_in[5];
    const float* W3  = (const float*)d_in[6];
    const float* b3  = (const float*)d_in[7];
    const float* W4  = (const float*)d_in[8];
    const float* b4  = (const float*)d_in[9];
    const float* Wc  = (const float*)d_in[10];
    const float* bc  = (const float*)d_in[11];
    float* out = (float*)d_out;

    char* p = (char*)d_ws;
    auto take = [&](size_t b) -> char* { char* r = p; p += (b + 255) & ~(size_t)255; return r; };
    int*   degIn   = (int*)take((size_t)NNODES * 4);
    int*   degOut  = (int*)take((size_t)NNODES * 4);
    float* normSrc = (float*)take((size_t)NNODES * 4);
    float* normDst = (float*)take((size_t)NNODES * 4);
    int*   offsets = (int*)take((size_t)(NNODES + 1) * 4);
    int*   cursor  = (int*)take((size_t)NNODES * 4);
    int*   blkSums = (int*)take(512 * 4);
    int*   csr     = (int*)take((size_t)NEDGES * 4);
    unsigned short* hbuf = (unsigned short*)take((size_t)NNODES * 512 * 2);
    unsigned short* tbuf = (unsigned short*)take((size_t)NNODES * 512 * 2);
    unsigned short* Wt2  = (unsigned short*)take((size_t)512 * 512 * 2);
    unsigned short* Wt3  = (unsigned short*)take((size_t)256 * 512 * 2);
    unsigned short* Wt4  = (unsigned short*)take((size_t)128 * 256 * 2);
    float* pooled = (float*)take(128 * 4);

    (void)hipMemsetAsync(degIn,  0, (size_t)NNODES * 4, stream);
    (void)hipMemsetAsync(degOut, 0, (size_t)NNODES * 4, stream);
    (void)hipMemsetAsync(pooled, 0, 128 * 4, stream);

    const int NB = (NNODES + 255) / 256;           // 391
    const int EB = (NEDGES + 255) / 256;

    k_degrees<<<EB, 256, 0, stream>>>(src, dst, degIn, degOut);
    k_block_sums<<<NB, 256, 0, stream>>>(degIn, blkSums, NNODES);
    k_scan_sums<<<1, 512, 0, stream>>>(blkSums, NB);
    k_scan_final<<<NB, 256, 0, stream>>>(degIn, blkSums, offsets, NNODES);
    k_copy_i32<<<NB, 256, 0, stream>>>(offsets, cursor, NNODES);
    k_fill_csr<<<EB, 256, 0, stream>>>(src, dst, cursor, csr);
    k_norms<<<NB, 256, 0, stream>>>(degIn, degOut, normSrc, normDst);

    k_transpose<<<(512 * 512 + 255) / 256, 256, 0, stream>>>(W2, Wt2, 512, 512);
    k_transpose<<<(512 * 256 + 255) / 256, 256, 0, stream>>>(W3, Wt3, 512, 256);
    k_transpose<<<(256 * 128 + 255) / 256, 256, 0, stream>>>(W4, Wt4, 256, 128);

    const int nPairs = NNODES / 32;                // 3125 (exact: 100000 = 3125*32)
    const int blocksPerStrip = (nPairs + 7) / 8;   // 391

    // layer 1 (outer product) -> t ; aggregate -> h (512)
    k_layer1<<<(int)(((size_t)NNODES * 512 + 255) / 256), 256, 0, stream>>>(degIn, normSrc, W1, tbuf);
    k_aggregate<<<NNODES, 256, 0, stream>>>(tbuf, csr, offsets, normDst, b1, hbuf, 512);

    // layer 2: h(512) @ W2(512x512)
    k_gemm<<<(512 / 64) * blocksPerStrip, 256, 0, stream>>>(hbuf, Wt2, normSrc, tbuf, 512, 512, nPairs, blocksPerStrip);
    k_aggregate<<<NNODES, 256, 0, stream>>>(tbuf, csr, offsets, normDst, b2, hbuf, 512);

    // layer 3: h(512) @ W3(512x256)
    k_gemm<<<(256 / 64) * blocksPerStrip, 256, 0, stream>>>(hbuf, Wt3, normSrc, tbuf, 512, 256, nPairs, blocksPerStrip);
    k_aggregate<<<NNODES, 256, 0, stream>>>(tbuf, csr, offsets, normDst, b3, hbuf, 256);

    // layer 4: h(256) @ W4(256x128)
    k_gemm<<<(128 / 64) * blocksPerStrip, 256, 0, stream>>>(hbuf, Wt4, normSrc, tbuf, 256, 128, nPairs, blocksPerStrip);
    k_aggregate<<<NNODES, 256, 0, stream>>>(tbuf, csr, offsets, normDst, b4, hbuf, 128);

    k_pool<<<256, 128, 0, stream>>>(hbuf, pooled);
    k_classify<<<1, 32, 0, stream>>>(pooled, Wc, bc, out);
}
